// SuperKimiDeltaAttention_52553219833861
// MI455X (gfx1250) — matrix operand order repaired and checked by reading, not compiled, not run on hardware
//
#include <hip/hip_runtime.h>
#include <math.h>

// ---- problem constants ----
#define B_   2
#define T_   4096
#define D_   2048
#define H_   16
#define HK_  64
#define PS_  1024
#define NC_  64        // T/C
#define CC_  64        // chunk length
#define LAT_ 128
#define MR_  8192      // B*T
#define LDF  68        // f32 LDS stride (272B, 16B-multiple)

typedef __bf16 bf16_t;
typedef __attribute__((ext_vector_type(16))) __bf16 v16bf;
typedef __attribute__((ext_vector_type(8)))  __bf16 v8bf;
typedef __attribute__((ext_vector_type(4)))  __bf16 v4bf;
typedef __attribute__((ext_vector_type(8)))  float  v8f;
typedef __attribute__((ext_vector_type(4)))  float  v4f;
typedef __attribute__((ext_vector_type(4)))  unsigned int v4u;
typedef __attribute__((ext_vector_type(8)))  int v8i;
typedef __attribute__((ext_vector_type(4)))  int v4i;

// ================= WMMA fragment helpers (CDNA5 wave32, 16x16x32 bf16) =================
__device__ __forceinline__ v16bf frag_a_row(const bf16_t* base, int ld, int m0, int k0) {
  int lane = threadIdx.x & 31;
  const bf16_t* p = base + (size_t)(m0 + (lane & 15)) * ld + k0 + ((lane & 16) ? 8 : 0);
  v8bf lo = *(const v8bf*)p;
  v8bf hi = *(const v8bf*)(p + 16);
  v16bf a;
#pragma unroll
  for (int i = 0; i < 8; ++i) { a[i] = lo[i]; a[8 + i] = hi[i]; }
  return a;
}
__device__ __forceinline__ v16bf frag_b_nt(const bf16_t* baseT, int ld, int n0, int k0) {
  int lane = threadIdx.x & 31;
  const bf16_t* p = baseT + (size_t)(n0 + (lane & 15)) * ld + k0 + ((lane & 16) ? 8 : 0);
  v8bf lo = *(const v8bf*)p;
  v8bf hi = *(const v8bf*)(p + 16);
  v16bf b;
#pragma unroll
  for (int i = 0; i < 8; ++i) { b[i] = lo[i]; b[8 + i] = hi[i]; }
  return b;
}
__device__ __forceinline__ v8f wmma_bf16(v16bf a, v16bf b, v8f c) {
  return __builtin_amdgcn_wmma_f32_16x16x32_bf16(false, a, false, b, (short)0, c, false, false);
}
__device__ __forceinline__ float sigmoidf_(float x) { return 1.f / (1.f + expf(-x)); }
__device__ __forceinline__ void st_bf4(bf16_t* p, v4f v) {
  v4bf o;
#pragma unroll
  for (int j = 0; j < 4; ++j) o[j] = (bf16_t)v[j];
  *(v4bf*)p = o;
}

// ================= Tensor Data Mover: 2D tile -> LDS (D# built per ISA 08_async_tensor §8) =====
// dsz: 1=2B(bf16), 2=4B(f32). padi: pad after 2^(padi+1) DWORDs; pada: (pada+1) DWORDs of pad.
__device__ __forceinline__ unsigned lds_addr_of(const void* p) {
  return (unsigned)(size_t)p;           // generic LDS pointer low 32 bits = LDS byte address
}
__device__ __forceinline__ void tdm_load_2d(
    unsigned lds_addr, const void* gptr, unsigned dsz,
    unsigned dim0, unsigned dim1, unsigned t0, unsigned t1,
    unsigned stride0, unsigned padi, unsigned pada)
{
  unsigned long long ga = (unsigned long long)gptr;
  v4u g0;
  g0[0] = 1u;                                               // count=1 (valid, user mode)
  g0[1] = lds_addr;                                         // lds_addr [63:32]
  g0[2] = (unsigned)ga;                                     // global_addr low
  g0[3] = ((unsigned)(ga >> 32) & 0x01FFFFFFu) | (2u << 30); // global_addr hi | type=2
  v8i g1;
  g1[0] = (int)((dsz << 16) | (1u << 20) | (padi << 22) | (pada << 25)); // data_size|pad_en|pad
  g1[1] = (int)((dim0 & 0xFFFFu) << 16);                    // tensor_dim0 [79:48] low half
  g1[2] = (int)((dim0 >> 16) | ((dim1 & 0xFFFFu) << 16));   // tensor_dim0 hi | tensor_dim1 lo
  g1[3] = (int)((dim1 >> 16) | (t0 << 16));                 // tensor_dim1 hi | tile_dim0
  g1[4] = (int)(t1 & 0xFFFFu);                              // tile_dim1 (tile_dim2=0)
  g1[5] = (int)stride0;                                     // tensor_dim0_stride low
  g1[6] = 0;
  g1[7] = 0;
  v4i z; z[0] = 0; z[1] = 0; z[2] = 0; z[3] = 0;
  v8i z8; z8[0] = 0; z8[1] = 0; z8[2] = 0; z8[3] = 0; z8[4] = 0; z8[5] = 0; z8[6] = 0; z8[7] = 0;
  __builtin_amdgcn_tensor_load_to_lds(g0, g1, z, z, z8, 0);
}
// TENSORcnt is per-wave: waves that issued nothing have it at 0, so an
// unconditional wait costs nothing and avoids exec-mask juggling.
__device__ __forceinline__ void tdm_wait() { __builtin_amdgcn_s_wait_tensorcnt((short)0); }

// ================= one-time converters =================
__global__ __launch_bounds__(256) void cvt_bf16_kernel(const float* __restrict__ in,
                                                       bf16_t* __restrict__ out, long n4) {
  long i = (long)blockIdx.x * 256 + threadIdx.x;
  if (i < n4) {
    v4f v = *(const v4f*)(in + i * 4);
    st_bf4(out + i * 4, v);
  }
}

__global__ __launch_bounds__(256) void transpose_bf16_kernel(
    const float* __restrict__ W, bf16_t* __restrict__ Wt, int K, int N)
{
  __shared__ bf16_t tile[32][33];
  int tx = threadIdx.x & 31, ty = threadIdx.x >> 5;
  int k0 = blockIdx.x * 32, n0 = blockIdx.y * 32;
#pragma unroll
  for (int j = 0; j < 4; ++j) {
    int r = k0 + ty + j * 8, c = n0 + tx;
    float v = (r < K && c < N) ? W[(size_t)r * N + c] : 0.f;
    tile[ty + j * 8][tx] = (bf16_t)v;
  }
  __syncthreads();
#pragma unroll
  for (int j = 0; j < 4; ++j) {
    int r = n0 + ty + j * 8, c = k0 + tx;
    if (r < N && c < K) Wt[(size_t)r * K + c] = tile[tx][ty + j * 8];
  }
}

// ================= bf16 GEMM: TDM-fed, double-buffered LDS, WMMA core =================
// C[M,N] = A[M,K] @ Wt[N,K]^T ; M%128==0, K%32==0; OOB rows (N<128 tiles) zero-filled by TDM.
__global__ __launch_bounds__(256) void gemm_bf16_kernel(
    const bf16_t* __restrict__ A, const bf16_t* __restrict__ Wt,
    float* __restrict__ Cm, bf16_t* __restrict__ Cbf,
    int M, int K, int N)
{
  __shared__ __align__(16) bf16_t As[2][128 * 40];
  __shared__ __align__(16) bf16_t Bs[2][128 * 40];
  int tid = threadIdx.x;
  int lane = tid & 31, wave = tid >> 5;
  int bm = blockIdx.y * 128, bn = blockIdx.x * 128;
  int wm = (wave >> 1) * 32, wn = (wave & 1) * 64;

  v8f acc[2][4];
#pragma unroll
  for (int im = 0; im < 2; ++im)
#pragma unroll
    for (int in = 0; in < 4; ++in)
#pragma unroll
      for (int v = 0; v < 8; ++v) acc[im][in][v] = 0.f;

  unsigned nrem = (unsigned)(N > bn ? N - bn : 0);
  if (wave == 0) {   // DMA first tiles; pad 4 DWORDs every 16 -> LDS row stride 40 bf16
    tdm_load_2d(lds_addr_of(As[0]), A + (size_t)bm * K, 1, (unsigned)K, (unsigned)(M - bm),
                32, 128, (unsigned)K, 3, 3);
    tdm_load_2d(lds_addr_of(Bs[0]), Wt + (size_t)bn * K, 1, (unsigned)K, nrem,
                32, 128, (unsigned)K, 3, 3);
  }
  tdm_wait();
  __syncthreads();

  int buf = 0;
  for (int k0 = 0; k0 < K; k0 += 32) {
    int kn = k0 + 32;
    if (kn < K && wave == 0) {   // DMA next slice while WMMAs run
      tdm_load_2d(lds_addr_of(As[buf ^ 1]), A + (size_t)bm * K + kn, 1,
                  (unsigned)(K - kn), (unsigned)(M - bm), 32, 128, (unsigned)K, 3, 3);
      tdm_load_2d(lds_addr_of(Bs[buf ^ 1]), Wt + (size_t)bn * K + kn, 1,
                  (unsigned)(K - kn), nrem, 32, 128, (unsigned)K, 3, 3);
    }
    const bf16_t* Ac = As[buf];
    const bf16_t* Bc = Bs[buf];
    v16bf bfr[4];
#pragma unroll
    for (int in = 0; in < 4; ++in) bfr[in] = frag_b_nt(Bc, 40, wn + in * 16, 0);
#pragma unroll
    for (int im = 0; im < 2; ++im) {
      v16bf af = frag_a_row(Ac, 40, wm + im * 16, 0);
#pragma unroll
      for (int in = 0; in < 4; ++in)
        acc[im][in] = wmma_bf16(af, bfr[in], acc[im][in]);
    }
    tdm_wait();
    __syncthreads();
    buf ^= 1;
  }
  // f32 epilogue (always)
#pragma unroll
  for (int im = 0; im < 2; ++im)
#pragma unroll
    for (int in = 0; in < 4; ++in) {
      int ncol = bn + wn + in * 16 + (lane & 15);
      int mrow = bm + wm + im * 16 + ((lane & 16) ? 8 : 0);
      if (ncol < N) {
#pragma unroll
        for (int v = 0; v < 8; ++v)
          Cm[(size_t)(mrow + v) * N + ncol] = acc[im][in][v];
      }
    }
  // optional bf16 epilogue (branch hoisted out of the store loops)
  if (Cbf) {
#pragma unroll
    for (int im = 0; im < 2; ++im)
#pragma unroll
      for (int in = 0; in < 4; ++in) {
        int ncol = bn + wn + in * 16 + (lane & 15);
        int mrow = bm + wm + im * 16 + ((lane & 16) ? 8 : 0);
        if (ncol < N) {
#pragma unroll
          for (int v = 0; v < 8; ++v)
            Cbf[(size_t)(mrow + v) * N + ncol] = (bf16_t)acc[im][in][v];
        }
      }
  }
}

// ================= conv(K=4)+SiLU + q/k norm + dt/log_g + beta sigmoid =================
__global__ __launch_bounds__(256) void conv_dt_kernel(
    const float* __restrict__ qpre, const float* __restrict__ kpre, const float* __restrict__ vpre,
    const float* __restrict__ qcw, const float* __restrict__ kcw, const float* __restrict__ vcw,
    const float* __restrict__ A_log, const float* __restrict__ dt_bias,
    float* __restrict__ qpost, float* __restrict__ kpost, float* __restrict__ vpost,
    float* __restrict__ g_logg, float* __restrict__ beta_io)
{
  __shared__ float rq[4][64], rk[4][64];
  int tid = threadIdx.x;
  int lh = tid >> 6, hk = tid & 63;
  size_t gh = (size_t)blockIdx.x * 4 + lh;
  int m = (int)(gh >> 4), h = (int)(gh & 15);
  int b = m >> 12, t = m & (T_ - 1);
  int p = h * HK_ + hk;
  size_t rowbase = (size_t)m * PS_ + p;

  float aq = 0.f, ak = 0.f, av = 0.f;
#pragma unroll
  for (int j = 0; j < 4; ++j) {
    int tt = t - 3 + j;
    if (tt >= 0) {
      size_t idx = ((size_t)(b * T_ + tt)) * PS_ + p;
      aq += qpre[idx] * qcw[p * 4 + j];
      ak += kpre[idx] * kcw[p * 4 + j];
      av += vpre[idx] * vcw[p * 4 + j];
    }
  }
  aq *= sigmoidf_(aq); ak *= sigmoidf_(ak); av *= sigmoidf_(av);
  rq[lh][hk] = aq * aq; rk[lh][hk] = ak * ak;
  __syncthreads();
  float sq = 0.f, sk = 0.f;
#pragma unroll 1
  for (int i = 0; i < 64; ++i) { sq += rq[lh][i]; sk += rk[lh][i]; }
  qpost[rowbase] = aq / (sqrtf(sq) + 1e-12f) * 0.125f;
  kpost[rowbase] = ak / (sqrtf(sk) + 1e-12f);
  vpost[rowbase] = av;

  float graw = g_logg[rowbase] + dt_bias[p];
  float sp = (graw > 20.f) ? graw : log1pf(expf(graw));
  g_logg[rowbase] = -expf(A_log[h]) * sp;

  if (hk == 0) {
    size_t bi = (size_t)m * H_ + h;
    beta_io[bi] = sigmoidf_(beta_io[bi]);
  }
}

// ================= per-chunk prep: emits bf16 operand tensors for the scan =================
__global__ __launch_bounds__(256) void chunk_prep_kernel(
    const float* __restrict__ qpost, const float* __restrict__ kpost, const float* __restrict__ vpost,
    const float* __restrict__ logg, const float* __restrict__ beta,
    bf16_t* __restrict__ utb_g /* transposed [k][c] */, bf16_t* __restrict__ ewb_g,
    float* __restrict__ effv_g, bf16_t* __restrict__ qtb_g, bf16_t* __restrict__ atb_g,
    float* __restrict__ wlast_g)
{
  __shared__ __align__(16) bf16_t u_bf[CC_ * 72];
  __shared__ __align__(16) bf16_t bw_bf[CC_ * 72];
  __shared__ __align__(16) bf16_t qt_bf[CC_ * 72];
  __shared__ __align__(16) float Lbuf[CC_ * LDF];
  __shared__ __align__(16) float bv_f[CC_ * LDF];
  __shared__ __align__(16) float bw_f[CC_ * LDF];

  int tid = threadIdx.x;
  int lane = tid & 31, wave = tid >> 5;
  int cid = blockIdx.x;
  int n = cid & (NC_ - 1);
  int h = (cid >> 6) & (H_ - 1);
  int b = cid >> 10;
  size_t row0 = (size_t)b * T_ + (size_t)n * CC_;
  size_t ch = (size_t)cid * (CC_ * HK_);

#pragma unroll
  for (int j = 0; j < 4; ++j) {
    int g4 = tid + j * 256;
    int i = g4 >> 4, kk4 = (g4 & 15) * 4;
    v4f v = *(const v4f*)(logg + (row0 + i) * PS_ + h * HK_ + kk4);
    *(v4f*)(Lbuf + i * LDF + kk4) = v;
  }
  __syncthreads();
  if (tid < 64) {
    float run = 0.f;
#pragma unroll 1
    for (int i = 0; i < CC_; ++i) {
      run += Lbuf[i * LDF + tid];
      Lbuf[i * LDF + tid] = fmaxf(run, -15.f);
    }
  }
  __syncthreads();
#pragma unroll
  for (int j = 0; j < 4; ++j) {
    int g4 = tid + j * 256;
    int i = g4 >> 4, kk4 = (g4 & 15) * 4;
    size_t g = (row0 + i) * PS_ + h * HK_ + kk4;
    v4f cum = *(const v4f*)(Lbuf + i * LDF + kk4);
    v4f kv = *(const v4f*)(kpost + g);
    v4f qv = *(const v4f*)(qpost + g);
    v4f vv = *(const v4f*)(vpost + g);
    float bt = beta[(row0 + i) * H_ + h];
    v4f uu, bw, qt, bv, wd4;
#pragma unroll
    for (int e = 0; e < 4; ++e) {
      float wd = expf(cum[e]), wi = expf(-cum[e]);
      uu[e] = kv[e] * wi; bw[e] = bt * kv[e] * wd;
      qt[e] = qv[e] * wd; bv[e] = bt * vv[e];
      wd4[e] = wd;
    }
#pragma unroll
    for (int e = 0; e < 4; ++e) utb_g[ch + (size_t)(kk4 + e) * CC_ + i] = (bf16_t)uu[e];
    st_bf4(qtb_g + ch + (size_t)i * HK_ + kk4, qt);
    st_bf4(u_bf + i * 72 + kk4, uu);
    st_bf4(qt_bf + i * 72 + kk4, qt);
    st_bf4(bw_bf + i * 72 + kk4, bw);
    *(v4f*)(bw_f + i * LDF + kk4) = bw;
    *(v4f*)(bv_f + i * LDF + kk4) = bv;
    if (i == CC_ - 1) *(v4f*)(wlast_g + (size_t)cid * HK_ + kk4) = wd4;
  }
  __syncthreads();
  // L = bw @ u^T (strictly lower); intra = qt @ u^T (lower incl diag)
#pragma unroll
  for (int tt = 0; tt < 2; ++tt) {
    int t = wave * 2 + tt;
    int m0 = (t >> 2) * 16, n0 = (t & 3) * 16;
    v8f accL, accI;
#pragma unroll
    for (int v = 0; v < 8; ++v) { accL[v] = 0.f; accI[v] = 0.f; }
#pragma unroll
    for (int k0 = 0; k0 < HK_; k0 += 32) {
      v16bf bu = frag_b_nt(u_bf, 72, n0, k0);
      accL = wmma_bf16(frag_a_row(bw_bf, 72, m0, k0), bu, accL);
      accI = wmma_bf16(frag_a_row(qt_bf, 72, m0, k0), bu, accI);
    }
    int ncol = n0 + (lane & 15);
    int mbase = m0 + ((lane & 16) ? 8 : 0);
#pragma unroll
    for (int v = 0; v < 8; ++v) {
      int mrow = mbase + v;
      Lbuf[mrow * LDF + ncol] = (ncol < mrow) ? accL[v] : 0.f;
      atb_g[ch + (size_t)mrow * HK_ + ncol] = (bf16_t)((ncol <= mrow) ? accI[v] : 0.f);
    }
  }
  __syncthreads();
  // (I+L)x = rhs forward substitution; 128 independent columns
  if (tid < 128) {
    float* rhs = (tid < 64) ? bv_f : bw_f;
    int cc = tid & 63;
#pragma unroll 1
    for (int i = 1; i < CC_; ++i) {
      float a = rhs[i * LDF + cc];
#pragma unroll 1
      for (int j = 0; j < i; ++j)
        a -= Lbuf[i * LDF + j] * rhs[j * LDF + cc];
      rhs[i * LDF + cc] = a;
    }
  }
  __syncthreads();
#pragma unroll
  for (int j = 0; j < 4; ++j) {
    int g4 = tid + j * 256;
    int i = g4 >> 4, kk4 = (g4 & 15) * 4;
    *(v4f*)(effv_g + ch + (size_t)i * HK_ + kk4) = *(const v4f*)(bv_f + i * LDF + kk4);
    st_bf4(ewb_g + ch + (size_t)i * HK_ + kk4, *(const v4f*)(bw_f + i * LDF + kk4));
  }
}

// ================= sequential chunk scan: TDM double-buffered operand feed =================
__global__ __launch_bounds__(256) void scan_kernel(
    const bf16_t* __restrict__ utb_g, const bf16_t* __restrict__ ewb_g,
    const float* __restrict__ effv_g, const bf16_t* __restrict__ qtb_g,
    const bf16_t* __restrict__ atb_g, const float* __restrict__ wlast_g,
    const float* __restrict__ cnw, const float* __restrict__ cdk,
    float* __restrict__ o_buf, float* __restrict__ snaps_out)
{
  __shared__ float S[HK_ * 65];
  __shared__ __align__(16) float ev_f[2][CC_ * LDF];
  __shared__ __align__(16) bf16_t S_t[HK_ * 72];
  __shared__ __align__(16) bf16_t d_t[HK_ * 72];
  __shared__ __align__(16) bf16_t ew_bf[2][CC_ * 72];
  __shared__ __align__(16) bf16_t qt_bf[2][CC_ * 72];
  __shared__ __align__(16) bf16_t at_bf[2][CC_ * 72];
  __shared__ __align__(16) bf16_t ut_bf[2][HK_ * 72];
  __shared__ float wl_s[HK_];
  __shared__ float red[256];

  int tid = threadIdx.x;
  int lane = tid & 31, wave = tid >> 5;
  int bh = blockIdx.x;
  int h = bh & (H_ - 1), b = bh >> 4;

#pragma unroll
  for (int j = 0; j < 16; ++j) {
    int idx = tid + j * 256;
    int i = idx >> 6, kk = idx & 63;
    S[i * 65 + kk] = 0.f;
    S_t[i * 72 + kk] = (bf16_t)0.f;
  }
  if (wave == 0) {   // DMA chunk 0 operands; bf16 tiles pad 4dw/32dw -> stride 72; f32 4dw/64dw -> 68
    size_t ch0 = (size_t)bh * NC_ * (CC_ * HK_);
    tdm_load_2d(lds_addr_of(ew_bf[0]), ewb_g + ch0, 1, 64, 64, 64, 64, 64, 4, 3);
    tdm_load_2d(lds_addr_of(qt_bf[0]), qtb_g + ch0, 1, 64, 64, 64, 64, 64, 4, 3);
    tdm_load_2d(lds_addr_of(at_bf[0]), atb_g + ch0, 1, 64, 64, 64, 64, 64, 4, 3);
    tdm_load_2d(lds_addr_of(ut_bf[0]), utb_g + ch0, 1, 64, 64, 64, 64, 64, 4, 3);
    tdm_load_2d(lds_addr_of(ev_f[0]),  effv_g + ch0, 2, 64, 64, 64, 64, 64, 5, 3);
  }
  tdm_wait();
  __syncthreads();

  int buf = 0;
  for (int n = 0; n < NC_; ++n) {
    if (wave == 0 && n + 1 < NC_) {   // DMA next chunk while this one computes
      size_t chn = ((size_t)bh * NC_ + n + 1) * (CC_ * HK_);
      int nb = buf ^ 1;
      tdm_load_2d(lds_addr_of(ew_bf[nb]), ewb_g + chn, 1, 64, 64, 64, 64, 64, 4, 3);
      tdm_load_2d(lds_addr_of(qt_bf[nb]), qtb_g + chn, 1, 64, 64, 64, 64, 64, 4, 3);
      tdm_load_2d(lds_addr_of(at_bf[nb]), atb_g + chn, 1, 64, 64, 64, 64, 64, 4, 3);
      tdm_load_2d(lds_addr_of(ut_bf[nb]), utb_g + chn, 1, 64, 64, 64, 64, 64, 4, 3);
      tdm_load_2d(lds_addr_of(ev_f[nb]),  effv_g + chn, 2, 64, 64, 64, 64, 64, 5, 3);
    }
    if (tid < HK_) wl_s[tid] = wlast_g[((size_t)bh * NC_ + n) * HK_ + tid];

    // delta = ev - ew @ S  -> d_t (transposed bf16)
#pragma unroll
    for (int tt = 0; tt < 2; ++tt) {
      int t = wave * 2 + tt;
      int m0 = (t >> 2) * 16, n0 = (t & 3) * 16;
      v8f acc;
#pragma unroll
      for (int v = 0; v < 8; ++v) acc[v] = 0.f;
#pragma unroll
      for (int k0 = 0; k0 < HK_; k0 += 32)
        acc = wmma_bf16(frag_a_row(ew_bf[buf], 72, m0, k0), frag_b_nt(S_t, 72, n0, k0), acc);
      int vcol = n0 + (lane & 15);
      int mbase = m0 + ((lane & 16) ? 8 : 0);
      v8bf db;
#pragma unroll
      for (int v = 0; v < 8; ++v)
        db[v] = (bf16_t)(ev_f[buf][(mbase + v) * LDF + vcol] - acc[v]);
      *(v8bf*)(d_t + vcol * 72 + mbase) = db;
    }
    __syncthreads();

    // o = qt @ S + at @ delta
#pragma unroll
    for (int tt = 0; tt < 2; ++tt) {
      int t = wave * 2 + tt;
      int m0 = (t >> 2) * 16, n0 = (t & 3) * 16;
      v8f acc;
#pragma unroll
      for (int v = 0; v < 8; ++v) acc[v] = 0.f;
#pragma unroll
      for (int k0 = 0; k0 < HK_; k0 += 32)
        acc = wmma_bf16(frag_a_row(qt_bf[buf], 72, m0, k0), frag_b_nt(S_t, 72, n0, k0), acc);
#pragma unroll
      for (int k0 = 0; k0 < CC_; k0 += 32)
        acc = wmma_bf16(frag_a_row(at_bf[buf], 72, m0, k0), frag_b_nt(d_t, 72, n0, k0), acc);
      int vcol = n0 + (lane & 15);
      int mbase = m0 + ((lane & 16) ? 8 : 0);
#pragma unroll
      for (int v = 0; v < 8; ++v)
        o_buf[((size_t)b * T_ + (size_t)n * CC_ + mbase + v) * PS_ + h * HK_ + vcol] = acc[v];
    }
    __syncthreads();

    // S = wl_row * (S + u^T @ delta)
#pragma unroll
    for (int tt = 0; tt < 2; ++tt) {
      int t = wave * 2 + tt;
      int m0 = (t >> 2) * 16, n0 = (t & 3) * 16;
      int vcol = n0 + (lane & 15);
      int mbase = m0 + ((lane & 16) ? 8 : 0);
      v8f acc;
#pragma unroll
      for (int v = 0; v < 8; ++v) acc[v] = S[(mbase + v) * 65 + vcol];
#pragma unroll
      for (int k0 = 0; k0 < CC_; k0 += 32)
        acc = wmma_bf16(frag_a_row(ut_bf[buf], 72, m0, k0), frag_b_nt(d_t, 72, n0, k0), acc);
      v8bf sb;
#pragma unroll
      for (int v = 0; v < 8; ++v) {
        int kr = mbase + v;
        float sv = wl_s[kr] * acc[v];
        S[kr * 65 + vcol] = sv;
        sb[v] = (bf16_t)sv;
      }
      *(v8bf*)(S_t + vcol * 72 + mbase) = sb;
    }
    tdm_wait();
    __syncthreads();

    if ((n & 3) == 3) {
      float ss = 0.f;
#pragma unroll
      for (int j = 0; j < 16; ++j) {
        int idx = tid + j * 256;
        float v = S[(idx >> 6) * 65 + (idx & 63)];
        ss += v * v;
      }
      red[tid] = ss;
      __syncthreads();
      for (int s = 128; s > 0; s >>= 1) {
        if (tid < s) red[tid] += red[tid + s];
        __syncthreads();
      }
      float rr = rsqrtf(red[0] * (1.f / 4096.f) + 1e-6f);
      if (tid < LAT_) {
        float acc = 0.f;
#pragma unroll 1
        for (int p = 0; p < 4096; ++p)
          acc += S[(p >> 6) * 65 + (p & 63)] * cnw[p] * cdk[(size_t)p * LAT_ + tid];
        int snap = n >> 2;
        snaps_out[(((size_t)b * (NC_ / 4) + snap) * H_ + h) * LAT_ + tid] = acc * rr;
      }
      __syncthreads();
    }
    buf ^= 1;
  }
}

// ================= head RMSNorm + gate; emits bf16 activation for final GEMM =================
__global__ __launch_bounds__(256) void onorm_kernel(
    const float* __restrict__ o_in, bf16_t* __restrict__ o_bf,
    const float* __restrict__ gate_raw, const float* __restrict__ g_b_b,
    const float* __restrict__ o_norm_w)
{
  __shared__ float rs[4][64];
  int tid = threadIdx.x;
  int lh = tid >> 6, hk = tid & 63;
  size_t gh = (size_t)blockIdx.x * 4 + lh;
  int m = (int)(gh >> 4), h = (int)(gh & 15);
  int p = h * HK_ + hk;
  size_t rowbase = (size_t)m * PS_ + p;
  float ov = o_in[rowbase];
  rs[lh][hk] = ov * ov;
  __syncthreads();
  float s = 0.f;
#pragma unroll 1
  for (int i = 0; i < 64; ++i) s += rs[lh][i];
  float rr = rsqrtf(s * (1.f / 64.f) + 1e-6f);
  float g = gate_raw[rowbase] + g_b_b[p];
  o_bf[rowbase] = (bf16_t)(ov * rr * o_norm_w[hk] * sigmoidf_(g));
}

__global__ void pos_kernel(int* __restrict__ out) {
  int i = threadIdx.x;
  if (i < 16) out[i] = (i + 1) * 256 - 1;
}

// ================================ host launch ================================
extern "C" void kernel_launch(void* const* d_in, const int* in_sizes, int n_in,
                              void* d_out, int out_size, void* d_ws, size_t ws_size,
                              hipStream_t stream)
{
  const float* x    = (const float*)d_in[0];
  const float* qw   = (const float*)d_in[1];
  const float* kw   = (const float*)d_in[2];
  const float* vw   = (const float*)d_in[3];
  const float* qcw  = (const float*)d_in[4];
  const float* kcw  = (const float*)d_in[5];
  const float* vcw  = (const float*)d_in[6];
  const float* Alog = (const float*)d_in[7];
  const float* dtb  = (const float*)d_in[8];
  const float* faw  = (const float*)d_in[9];
  const float* fbw  = (const float*)d_in[10];
  const float* bpw  = (const float*)d_in[11];
  const float* gaw  = (const float*)d_in[12];
  const float* gbw  = (const float*)d_in[13];
  const float* gbb  = (const float*)d_in[14];
  const float* onw  = (const float*)d_in[15];
  const float* opw  = (const float*)d_in[16];
  const float* cnw  = (const float*)d_in[17];
  const float* cdk  = (const float*)d_in[18];

  const size_t MPS = (size_t)MR_ * PS_;
  const size_t MH  = (size_t)MR_ * H_;
  const size_t MHK = (size_t)MR_ * HK_;
  const size_t MD  = (size_t)MR_ * D_;
  const size_t CHT = (size_t)B_ * H_ * NC_ * CC_ * HK_;

  float* ws = (float*)d_ws;
  size_t off = 0;
  float* q_pre   = ws + off; off += MPS;
  float* k_pre   = ws + off; off += MPS;
  float* v_pre   = ws + off; off += MPS;
  float* q_post  = ws + off; off += MPS;
  float* k_post  = ws + off; off += MPS;
  float* v_post  = ws + off; off += MPS;
  float* g_low   = ws + off; off += MHK;
  float* gt_low  = ws + off; off += MHK;
  float* g_raw   = ws + off; off += MPS;   // -> log_g in place
  float* gt_raw  = ws + off; off += MPS;
  float* beta_b  = ws + off; off += MH;
  float* effv_g  = ws + off; off += CHT;
  float* wlast   = ws + off; off += (size_t)B_ * H_ * NC_ * HK_;
  float* o_buf   = ws + off; off += MPS;

  bf16_t* bws = (bf16_t*)(ws + off);
  size_t boff = 0;
  bf16_t* x_bf    = bws + boff; boff += MD;
  bf16_t* qwt     = bws + boff; boff += (size_t)PS_ * D_;
  bf16_t* kwt     = bws + boff; boff += (size_t)PS_ * D_;
  bf16_t* vwt     = bws + boff; boff += (size_t)PS_ * D_;
  bf16_t* opwt    = bws + boff; boff += (size_t)D_ * PS_;
  bf16_t* fawt    = bws + boff; boff += (size_t)HK_ * D_;
  bf16_t* gawt    = bws + boff; boff += (size_t)HK_ * D_;
  bf16_t* fbwt    = bws + boff; boff += (size_t)PS_ * HK_;
  bf16_t* gbwt    = bws + boff; boff += (size_t)PS_ * HK_;
  bf16_t* bpwt    = bws + boff; boff += (size_t)H_ * D_;
  bf16_t* glow_bf = bws + boff; boff += MHK;
  bf16_t* gtlow_bf= bws + boff; boff += MHK;
  bf16_t* o_bf    = bws + boff; boff += MPS;
  bf16_t* utb_g   = bws + boff; boff += CHT;
  bf16_t* ewb_g   = bws + boff; boff += CHT;
  bf16_t* qtb_g   = bws + boff; boff += CHT;
  bf16_t* atb_g   = bws + boff; boff += CHT;

  dim3 blk(256);

  cvt_bf16_kernel<<<dim3((unsigned)(MD / 4 / 256)), blk, 0, stream>>>(x, x_bf, (long)(MD / 4));
  transpose_bf16_kernel<<<dim3(D_ / 32, PS_ / 32), blk, 0, stream>>>(qw, qwt, D_, PS_);
  transpose_bf16_kernel<<<dim3(D_ / 32, PS_ / 32), blk, 0, stream>>>(kw, kwt, D_, PS_);
  transpose_bf16_kernel<<<dim3(D_ / 32, PS_ / 32), blk, 0, stream>>>(vw, vwt, D_, PS_);
  transpose_bf16_kernel<<<dim3(PS_ / 32, D_ / 32), blk, 0, stream>>>(opw, opwt, PS_, D_);
  transpose_bf16_kernel<<<dim3(D_ / 32, HK_ / 32), blk, 0, stream>>>(faw, fawt, D_, HK_);
  transpose_bf16_kernel<<<dim3(D_ / 32, HK_ / 32), blk, 0, stream>>>(gaw, gawt, D_, HK_);
  transpose_bf16_kernel<<<dim3(HK_ / 32, PS_ / 32), blk, 0, stream>>>(fbw, fbwt, HK_, PS_);
  transpose_bf16_kernel<<<dim3(HK_ / 32, PS_ / 32), blk, 0, stream>>>(gbw, gbwt, HK_, PS_);
  transpose_bf16_kernel<<<dim3(D_ / 32, 1), blk, 0, stream>>>(bpw, bpwt, D_, H_);

  dim3 gBig(PS_ / 128, MR_ / 128);
  gemm_bf16_kernel<<<gBig, blk, 0, stream>>>(x_bf, qwt, q_pre, nullptr, MR_, D_, PS_);
  gemm_bf16_kernel<<<gBig, blk, 0, stream>>>(x_bf, kwt, k_pre, nullptr, MR_, D_, PS_);
  gemm_bf16_kernel<<<gBig, blk, 0, stream>>>(x_bf, vwt, v_pre, nullptr, MR_, D_, PS_);
  gemm_bf16_kernel<<<dim3(1, MR_ / 128), blk, 0, stream>>>(x_bf, fawt, g_low, glow_bf, MR_, D_, HK_);
  gemm_bf16_kernel<<<gBig, blk, 0, stream>>>(glow_bf, fbwt, g_raw, nullptr, MR_, HK_, PS_);
  gemm_bf16_kernel<<<dim3(1, MR_ / 128), blk, 0, stream>>>(x_bf, gawt, gt_low, gtlow_bf, MR_, D_, HK_);
  gemm_bf16_kernel<<<gBig, blk, 0, stream>>>(gtlow_bf, gbwt, gt_raw, nullptr, MR_, HK_, PS_);
  gemm_bf16_kernel<<<dim3(1, MR_ / 128), blk, 0, stream>>>(x_bf, bpwt, beta_b, nullptr, MR_, D_, H_);

  conv_dt_kernel<<<dim3(MR_ * H_ / 4), blk, 0, stream>>>(
      q_pre, k_pre, v_pre, qcw, kcw, vcw, Alog, dtb,
      q_post, k_post, v_post, g_raw, beta_b);

  chunk_prep_kernel<<<dim3(B_ * H_ * NC_), blk, 0, stream>>>(
      q_post, k_post, v_post, g_raw, beta_b,
      utb_g, ewb_g, effv_g, qtb_g, atb_g, wlast);

  float* out_f   = (float*)d_out;
  float* snaps_f = out_f + MD;
  scan_kernel<<<dim3(B_ * H_), blk, 0, stream>>>(
      utb_g, ewb_g, effv_g, qtb_g, atb_g, wlast, cnw, cdk, o_buf, snaps_f);

  onorm_kernel<<<dim3(MR_ * H_ / 4), blk, 0, stream>>>(o_buf, o_bf, gt_raw, gbb, onw);

  gemm_bf16_kernel<<<dim3(D_ / 128, MR_ / 128), blk, 0, stream>>>(o_bf, opwt, out_f, nullptr, MR_, PS_, D_);

  int* pos = (int*)(snaps_f + (size_t)B_ * (NC_ / 4) * H_ * LAT_);
  pos_kernel<<<1, 16, 0, stream>>>(pos);

  (void)in_sizes; (void)n_in; (void)out_size; (void)ws_size;
}
